// LemInFrame_85744727097803
// MI455X (gfx1250) — compile-verified, matrix operand order checked by reading, not run on hardware
//
#include <hip/hip_runtime.h>
#include <hip/hip_bf16.h>
#include <math.h>

// ---------------------------------------------------------------------------
// MI455X (gfx1250) implementation. wave32, WMMA bf16 16x16x32, f32 accum.
// One wave = one 16-atom tile. Weights are pre-swizzled once per launch into
// bf16 WMMA B-fragment layout in d_ws (so the hot loop loads 32B/lane packed
// fragments). Segment-sum via f32 global atomics, then a finalize kernel
// applies the 9x9 Q matrix + cartesian permutation.
// ---------------------------------------------------------------------------

typedef __bf16 bf16;
typedef __attribute__((ext_vector_type(16))) __bf16 v16bf;
typedef __attribute__((ext_vector_type(8)))  float  v8f;

#define DEVINL __device__ __forceinline__

// K index of element e (0..15) of a 16x32 bf16 A/B fragment for lane-half hi.
// Per ISA 7.12.2: vgpr j<4 -> K=2j(+1), j>=4 -> K=16+2(j-4)(+1); hi adds 8.
DEVINL constexpr int frag_k(int e, int hi) {
  int j2 = e >> 1, p = e & 1;
  int base = (j2 < 4) ? (2 * j2) : (16 + 2 * (j2 - 4));
  return base + 8 * hi + p;
}

// A fragment, unit-stride source: pairs are contiguous -> float2 loads feeding
// packed f32->bf16 conversion (v_cvt_pk_bf16_f32).
DEVINL v16bf load_a1(const float* __restrict__ rowPtr, int hi, int kOff) {
  v16bf a;
#pragma unroll
  for (int j2 = 0; j2 < 8; ++j2) {
    int k = ((j2 < 4) ? (2 * j2) : (16 + 2 * (j2 - 4))) + 8 * hi + kOff;
    float2 f = *(const float2*)(rowPtr + k);
    a[2 * j2] = (bf16)f.x;
    a[2 * j2 + 1] = (bf16)f.y;
  }
  return a;
}

// A fragment, strided source (spherical l=1/l=2 component gathers)
DEVINL v16bf load_as(const float* __restrict__ rowPtr, int hi, int kOff,
                     int stride, int cOff) {
  v16bf a;
#pragma unroll
  for (int e = 0; e < 16; ++e)
    a[e] = (bf16)rowPtr[cOff + stride * (kOff + frag_k(e, hi))];
  return a;
}

// A fragment gather from LDS bf16 tile [16 x 64]
DEVINL v16bf load_a_lds(const bf16* __restrict__ h, int row, int hi, int kOff) {
  v16bf a;
#pragma unroll
  for (int e = 0; e < 16; ++e) a[e] = h[row * 64 + kOff + frag_k(e, hi)];
  return a;
}

// Pre-swizzled B fragment: 512 bf16 per fragment, 16 contiguous per lane.
DEVINL v16bf load_bw(const bf16* __restrict__ base, int frag, int lane) {
  return ((const v16bf*)(base + (size_t)frag * 512))[lane];
}

DEVINL v8f wmma_bf16(v16bf a, v16bf b, v8f c) {
  return __builtin_amdgcn_wmma_f32_16x16x32_bf16(
      /*neg_a=*/false, a, /*neg_b=*/false, b,
      /*c_mod=*/(short)0, c, /*reuse_a=*/false, /*reuse_b=*/false);
}

// ------------------------ Wigner-3j constant tables -------------------------
static constexpr float C1 = 0.31622776601683794f;  // 1/sqrt(10)
static constexpr float C2 = 0.18257418583505536f;  // 1/sqrt(30)
static constexpr float CG112[3][3][5] = {
    {{0, 0, -C2, 0, C1}, {C1, 0, 0, 0, 0}, {0, 0, 0, C1, 0}},
    {{C1, 0, 0, 0, 0}, {0, 0, 2 * C2, 0, 0}, {0, C1, 0, 0, 0}},
    {{0, 0, 0, C1, 0}, {0, C1, 0, 0, 0}, {0, 0, -C2, 0, -C1}}};

static constexpr float S30 = 0.18257418583505536f;  // 1/sqrt(30)
static constexpr float T30 = 0.31622776601683794f;  // sqrt(3)/sqrt(30)
// CG221[a][b][m] = (so3 generator G_m for l=2)/sqrt(30), antisymmetric in a,b
static constexpr float CG221[5][5][3] = {
    /*a=0*/ {{0, 0, 0}, {0, S30, 0}, {0, 0, 0}, {S30, 0, 0}, {0, 0, 2 * S30}},
    /*a=1*/ {{0, -S30, 0}, {0, 0, 0}, {T30, 0, 0}, {0, 0, S30}, {-S30, 0, 0}},
    /*a=2*/ {{0, 0, 0}, {-T30, 0, 0}, {0, 0, 0}, {0, -T30, 0}, {0, 0, 0}},
    /*a=3*/ {{-S30, 0, 0}, {0, 0, -S30}, {0, T30, 0}, {0, 0, 0}, {0, S30, 0}},
    /*a=4*/ {{0, 0, -2 * S30}, {S30, 0, 0}, {0, 0, 0}, {0, -S30, 0}, {0, 0, 0}}};

// CG222: fully symmetric d-symbols of the real l=2 rep, unit Frobenius norm.
DEVINL constexpr float cg222(int a, int b, int c) {
  int x = a, y = b, z = c, t;
  if (x > y) { t = x; x = y; y = t; }
  if (y > z) { t = y; y = z; z = t; }
  if (x > y) { t = x; x = y; y = t; }
  if (x == 0 && y == 1 && z == 3) return 0.2070196f;
  if (x == 0 && y == 0 && z == 2) return -0.2390457f;
  if (x == 1 && y == 1 && z == 2) return 0.1195229f;
  if (x == 2 && y == 3 && z == 3) return 0.1195229f;
  if (x == 2 && y == 4 && z == 4) return -0.2390457f;
  if (x == 2 && y == 2 && z == 2) return 0.2390457f;
  if (x == 1 && y == 1 && z == 4) return -0.2070196f;
  if (x == 3 && y == 3 && z == 4) return 0.2070196f;
  return 0.f;
}

// Q: 9x9 basis change (l0+l1+l2 spherical -> 3x3 cartesian flattened)
static constexpr float QM[9][9] = {
    {0.57735027f, 0, 0, 0, 0.57735027f, 0, 0, 0, 0.57735027f},
    {0, 0, 0, 0, 0, 0.70710678f, 0, -0.70710678f, 0},
    {0, 0, -0.70710678f, 0, 0, 0, 0.70710678f, 0, 0},
    {0, 0.70710678f, 0, -0.70710678f, 0, 0, 0, 0, 0},
    {0, 0.70710678f, 0, 0.70710678f, 0, 0, 0, 0, 0},
    {0, 0, 0, 0, 0, 0.70710678f, 0, 0.70710678f, 0},
    {-0.40824829f, 0, 0, 0, -0.40824829f, 0, 0, 0, 0.81649658f},
    {0, 0, 0.70710678f, 0, 0, 0, 0.70710678f, 0, 0},
    {0.70710678f, 0, 0, 0, -0.70710678f, 0, 0, 0, 0}};

// Weight-fragment workspace layout (bf16 elements, after ACC_BYTES offset)
#define WSOFF_GW1 0        // 4x4  frags
#define WSOFF_GW2 8192     // 9x2  frags
#define WSOFF_WU0 17408    // 4    frags
#define WSOFF_WV0 19456    // 4
#define WSOFF_WU1 21504    // 2
#define WSOFF_WV1 22528    // 2
#define WSOFF_WU2 23552    // 1
#define WSOFF_WV2 24064    // 1
#define ACC_BYTES 65536

// ---------------------------------------------------------------------------
__global__ void zero_kernel(float* __restrict__ p, int n) {
  int i = blockIdx.x * blockDim.x + threadIdx.x;
  if (i < n) p[i] = 0.f;
}

// Pre-swizzle one weight matrix [K x N] f32 into bf16 B-fragment layout.
// grid = nBlocks*kBlocks fragments, 32 threads each; frag = nb*kBlocks + kb.
__global__ void pack_b_kernel(const float* __restrict__ W, int ldn,
                              int kBlocks, bf16* __restrict__ dst) {
  int lane = threadIdx.x;
  int frag = blockIdx.x;
  int nb = frag / kBlocks, kb = frag % kBlocks;
  int n15 = lane & 15, hi = lane >> 4;
  bf16* out = dst + (size_t)frag * 512 + lane * 16;
  int nCol = nb * 16 + n15;
#pragma unroll
  for (int e = 0; e < 16; ++e)
    out[e] = (bf16)W[(kb * 32 + frag_k(e, hi)) * ldn + nCol];
}

__global__ __launch_bounds__(128) void tp_kernel(
    const int* __restrict__ batch, const float* __restrict__ xs,
    const float* __restrict__ xsph, const bf16* __restrict__ wb,
    const float* __restrict__ gb1, const float* __restrict__ gb2,
    const float* __restrict__ Wp0, const float* __restrict__ Wp1,
    const float* __restrict__ Wp2, float* __restrict__ acc, int nAtoms,
    int nTiles) {
  __shared__ bf16 hs[4][16 * 64];

  const int lane = threadIdx.x & 31;
  const int wid = threadIdx.x >> 5;
  const int n15 = lane & 15;
  const int hi = lane >> 4;

  int tileRaw = blockIdx.x * 4 + wid;
  bool writer = (tileRaw < nTiles);
  int tile = writer ? tileRaw : (nTiles - 1);
  int row0 = tile * 16;

  const float* xsRow = xs + (size_t)(row0 + n15) * 128;
  const float* sphRow = xsph + (size_t)(row0 + n15) * 480;

  // Prefetch next tile group's feature rows (global_prefetch_b8)
  if (tileRaw + 4 < nTiles) {
    __builtin_prefetch(xsph + (size_t)(row0 + 64 + n15) * 480, 0, 1);
    __builtin_prefetch(xs + (size_t)(row0 + 64 + n15) * 128, 0, 1);
  }

  // -------- GEMM1: h = silu(xs[16,128] @ gw1[128,64] + gb1) ---------------
#pragma unroll
  for (int nb = 0; nb < 4; ++nb) {
    float bias = gb1[nb * 16 + n15];
    v8f c = {bias, bias, bias, bias, bias, bias, bias, bias};
#pragma unroll
    for (int kb = 0; kb < 4; ++kb) {
      v16bf a = load_a1(xsRow, hi, kb * 32);
      v16bf b = load_bw(wb + WSOFF_GW1, nb * 4 + kb, lane);
      c = wmma_bf16(a, b, c);
    }
#pragma unroll
    for (int v = 0; v < 8; ++v) {
      float x = c[v];
      float s = x / (1.f + __expf(-x));  // silu
      hs[wid][(v + 8 * hi) * 64 + nb * 16 + n15] = (bf16)s;
    }
  }
  __syncthreads();

  // -------- GEMM2: w[16,144] = h[16,64] @ gw2[64,144] + gb2 ---------------
  // Column layout 16*k+u: D-fragment k has lane index == mul u, matching U/V.
  v16bf ha0 = load_a_lds(&hs[wid][0], n15, hi, 0);
  v16bf ha1 = load_a_lds(&hs[wid][0], n15, hi, 32);
  v8f wf[9];
#pragma unroll
  for (int k = 0; k < 9; ++k) {
    float bias = gb2[k * 16 + n15];
    v8f c = {bias, bias, bias, bias, bias, bias, bias, bias};
    c = wmma_bf16(ha0, load_bw(wb + WSOFF_GW2, k * 2 + 0, lane), c);
    c = wmma_bf16(ha1, load_bw(wb + WSOFF_GW2, k * 2 + 1, lane), c);
    wf[k] = c;
  }

  // -------- U/V irreps linears (A fragments shared between U and V) -------
  const float inv128 = 0.08838834764831845f;
  const float inv64 = 0.125f;
  const float inv32 = 0.17677669529663687f;

  v8f u0, v0, u1f[3], v1f[3], u2f[5], v2f[5];
  {
    v8f cu = {}; v8f cv = {};
#pragma unroll
    for (int kb = 0; kb < 4; ++kb) {
      v16bf a = load_a1(sphRow, hi, kb * 32);
      cu = wmma_bf16(a, load_bw(wb + WSOFF_WU0, kb, lane), cu);
      cv = wmma_bf16(a, load_bw(wb + WSOFF_WV0, kb, lane), cv);
    }
#pragma unroll
    for (int v = 0; v < 8; ++v) { cu[v] *= inv128; cv[v] *= inv128; }
    u0 = cu; v0 = cv;
  }
#pragma unroll
  for (int i = 0; i < 3; ++i) {
    v8f cu = {}; v8f cv = {};
#pragma unroll
    for (int kb = 0; kb < 2; ++kb) {
      v16bf a = load_as(sphRow, hi, kb * 32, 3, 128 + i);  // x1[:,u,i]
      cu = wmma_bf16(a, load_bw(wb + WSOFF_WU1, kb, lane), cu);
      cv = wmma_bf16(a, load_bw(wb + WSOFF_WV1, kb, lane), cv);
    }
#pragma unroll
    for (int v = 0; v < 8; ++v) { cu[v] *= inv64; cv[v] *= inv64; }
    u1f[i] = cu; v1f[i] = cv;
  }
#pragma unroll
  for (int i = 0; i < 5; ++i) {
    v16bf a = load_as(sphRow, hi, 0, 5, 320 + i);  // x2[:,u,i]
    v8f cu = {}; v8f cv = {};
    cu = wmma_bf16(a, load_bw(wb + WSOFF_WU2, 0, lane), cu);
    cv = wmma_bf16(a, load_bw(wb + WSOFF_WV2, 0, lane), cv);
#pragma unroll
    for (int v = 0; v < 8; ++v) { cu[v] *= inv32; cv[v] *= inv32; }
    u2f[i] = cu; v2f[i] = cv;
  }

  // -------- CG tensor-product epilogue -----------------------------------
  float wp0[3] = {Wp0[0], Wp0[1], Wp0[2]};
  float wp1[2] = {Wp1[0], Wp1[1]};
  float wp2[4] = {Wp2[0], Wp2[1], Wp2[2], Wp2[3]};
  const float i_s3 = 0.5773502691896258f;   // 1/sqrt(3)
  const float i_s5 = 0.4472135954999579f;   // 1/sqrt(5)
  const float s3_4 = 0.4330127018922193f;   // sqrt(3)/4
  const float s5_4 = 0.5590169943749475f;   // sqrt(5)/4

#pragma unroll
  for (int v = 0; v < 8; ++v) {
    float U0 = u0[v], V0 = v0[v];
    float U1[3] = {u1f[0][v], u1f[1][v], u1f[2][v]};
    float V1[3] = {v1f[0][v], v1f[1][v], v1f[2][v]};
    float U2[5] = {u2f[0][v], u2f[1][v], u2f[2][v], u2f[3][v], u2f[4][v]};
    float V2[5] = {v2f[0][v], v2f[1][v], v2f[2][v], v2f[3][v], v2f[4][v]};
    float w[9];
#pragma unroll
    for (int k = 0; k < 9; ++k) w[k] = wf[k][v];

    // lo = 0 paths (k = 0, 2, 6)
    float o0 = 0.25f * w[0] * U0 * V0;
    float d11 = U1[0] * V1[0] + U1[1] * V1[1] + U1[2] * V1[2];
    float o2 = 0.25f * i_s3 * w[2] * d11;
    float d22 = U2[0] * V2[0] + U2[1] * V2[1] + U2[2] * V2[2] +
                U2[3] * V2[3] + U2[4] * V2[4];
    float o6 = 0.25f * i_s5 * w[6] * d22;
    float y0 = (o0 * wp0[0] + o2 * wp0[1] + o6 * wp0[2]) * i_s3;

    // lo = 1 paths (k = 3: cross product, k = 7: generator contraction)
    float cr[3] = {U1[1] * V1[2] - U1[2] * V1[1],
                   U1[2] * V1[0] - U1[0] * V1[2],
                   U1[0] * V1[1] - U1[1] * V1[0]};
    float o7[3] = {0.f, 0.f, 0.f};
#pragma unroll
    for (int a = 0; a < 5; ++a)
#pragma unroll
      for (int b = 0; b < 5; ++b) {
        float uv = U2[a] * V2[b];
#pragma unroll
        for (int m = 0; m < 3; ++m)
          if (CG221[a][b][m] != 0.f) o7[m] += CG221[a][b][m] * uv;
      }
    float y1[3];
#pragma unroll
    for (int m = 0; m < 3; ++m) {
      float o3 = 0.17677669529663687f * w[3] * cr[m];  // sqrt3/(4 sqrt6)
      y1[m] = (o3 * wp1[0] + s3_4 * w[7] * o7[m] * wp1[1]) * 0.7071067811865475f;
    }

    // lo = 2 paths (k = 1, 4, 5, 8)
    float o4[5] = {0.f, 0.f, 0.f, 0.f, 0.f};
#pragma unroll
    for (int a = 0; a < 3; ++a)
#pragma unroll
      for (int b = 0; b < 3; ++b) {
        float uv = U1[a] * V1[b];
#pragma unroll
        for (int m = 0; m < 5; ++m)
          if (CG112[a][b][m] != 0.f) o4[m] += CG112[a][b][m] * uv;
      }
    float o8[5] = {0.f, 0.f, 0.f, 0.f, 0.f};
#pragma unroll
    for (int a = 0; a < 5; ++a)
#pragma unroll
      for (int b = 0; b < 5; ++b) {
        float uv = U2[a] * V2[b];
#pragma unroll
        for (int m = 0; m < 5; ++m)
          if (cg222(a, b, m) != 0.f) o8[m] += cg222(a, b, m) * uv;
      }
    float y2[5];
#pragma unroll
    for (int m = 0; m < 5; ++m) {
      float o1 = 0.25f * w[1] * U0 * V2[m];
      float o5 = 0.25f * w[5] * U2[m] * V0;
      y2[m] = (o1 * wp2[0] + s5_4 * w[4] * o4[m] * wp2[1] + o5 * wp2[2] +
               s5_4 * w[8] * o8[m] * wp2[3]) * 0.5f;
    }

    float out9[9] = {y0, y1[0], y1[1], y1[2], y2[0], y2[1], y2[2], y2[3], y2[4]};

    // Sum over mul u: butterfly across the 16 lanes sharing this half
#pragma unroll
    for (int j = 0; j < 9; ++j) {
      float s = out9[j];
      s += __shfl_xor(s, 1, 32);
      s += __shfl_xor(s, 2, 32);
      s += __shfl_xor(s, 4, 32);
      s += __shfl_xor(s, 8, 32);
      out9[j] = s;
    }

    if (writer && n15 == 0) {
      int atom = row0 + v + 8 * hi;
      if (atom < nAtoms) {
        int g = batch[atom];
#pragma unroll
        for (int j = 0; j < 9; ++j) atomicAdd(&acc[(size_t)g * 9 + j], out9[j]);
      }
    }
  }
}

__global__ void finalize_kernel(const float* __restrict__ acc,
                                float* __restrict__ out, int nGraphs) {
  int g = blockIdx.x * blockDim.x + threadIdx.x;
  if (g >= nGraphs) return;
  float s[9];
#pragma unroll
  for (int j = 0; j < 9; ++j) s[j] = acc[(size_t)g * 9 + j];
  float cart[9];
#pragma unroll
  for (int c = 0; c < 9; ++c) {
    float t = 0.f;
#pragma unroll
    for (int j = 0; j < 9; ++j)
      if (QM[j][c] != 0.f) t += s[j] * QM[j][c];
    cart[c] = t;
  }
  const int idx[3] = {2, 0, 1};
#pragma unroll
  for (int i = 0; i < 3; ++i)
#pragma unroll
    for (int j = 0; j < 3; ++j)
      out[(size_t)g * 9 + i * 3 + j] = cart[idx[i] * 3 + idx[j]];
}

extern "C" void kernel_launch(void* const* d_in, const int* in_sizes, int n_in,
                              void* d_out, int out_size, void* d_ws,
                              size_t ws_size, hipStream_t stream) {
  const int* batch = (const int*)d_in[0];
  const float* xs = (const float*)d_in[1];
  const float* xsph = (const float*)d_in[2];
  const float* WU0 = (const float*)d_in[3];
  const float* WU1 = (const float*)d_in[4];
  const float* WU2 = (const float*)d_in[5];
  const float* WV0 = (const float*)d_in[6];
  const float* WV1 = (const float*)d_in[7];
  const float* WV2 = (const float*)d_in[8];
  const float* gw1 = (const float*)d_in[9];
  const float* gb1 = (const float*)d_in[10];
  const float* gw2 = (const float*)d_in[11];
  const float* gb2 = (const float*)d_in[12];
  const float* Wp0 = (const float*)d_in[13];
  const float* Wp1 = (const float*)d_in[14];
  const float* Wp2 = (const float*)d_in[15];
  float* out = (float*)d_out;
  float* acc = (float*)d_ws;
  bf16* wb = (bf16*)((char*)d_ws + ACC_BYTES);

  int nAtoms = in_sizes[0];
  int nGraphs = out_size / 9;
  int nTiles = (nAtoms + 15) / 16;

  // Zero segment accumulators + pre-swizzle weights to bf16 fragments.
  zero_kernel<<<(nGraphs * 9 + 255) / 256, 256, 0, stream>>>(acc, nGraphs * 9);
  pack_b_kernel<<<16, 32, 0, stream>>>(gw1, 64, 4, wb + WSOFF_GW1);
  pack_b_kernel<<<18, 32, 0, stream>>>(gw2, 144, 2, wb + WSOFF_GW2);
  pack_b_kernel<<<4, 32, 0, stream>>>(WU0, 16, 4, wb + WSOFF_WU0);
  pack_b_kernel<<<4, 32, 0, stream>>>(WV0, 16, 4, wb + WSOFF_WV0);
  pack_b_kernel<<<2, 32, 0, stream>>>(WU1, 16, 2, wb + WSOFF_WU1);
  pack_b_kernel<<<2, 32, 0, stream>>>(WV1, 16, 2, wb + WSOFF_WV1);
  pack_b_kernel<<<1, 32, 0, stream>>>(WU2, 16, 1, wb + WSOFF_WU2);
  pack_b_kernel<<<1, 32, 0, stream>>>(WV2, 16, 1, wb + WSOFF_WV2);

  tp_kernel<<<(nTiles + 3) / 4, 128, 0, stream>>>(
      batch, xs, xsph, wb, gb1, gb2, Wp0, Wp1, Wp2, acc, nAtoms, nTiles);
  finalize_kernel<<<(nGraphs + 255) / 256, 256, 0, stream>>>(acc, out, nGraphs);
}
